// MS_Attention_RepConv_75368086110773
// MI455X (gfx1250) — matrix operand order repaired and checked by reading, not compiled, not run on hardware
//
#include <hip/hip_runtime.h>
#include <math.h>

// ---------------- constants ----------------
constexpr int T  = 4;
constexpr int Bz = 16;          // batch
constexpr int TB = T * Bz;      // 64 images
constexpr int C  = 512;
constexpr int C3 = 3 * C;       // 1536
constexpr int N  = 196;         // 14*14
constexpr int NP = 256;         // padded N (16 tiles of 16, 8 K-steps of 32)
constexpr int HEADS = 8;
constexpr int HD = 64;

typedef __attribute__((ext_vector_type(16))) _Float16 v16h;
typedef __attribute__((ext_vector_type(8)))  _Float16 v8h;
typedef __attribute__((ext_vector_type(8)))  float    v8f;
typedef int v4i __attribute__((vector_size(16)));   // matches builtin prototype

// ---- CDNA5 async global->LDS path (guarded: falls back to uint4 staging) ----
#if defined(__has_builtin)
#  if __has_builtin(__builtin_amdgcn_global_load_async_to_lds_b128) && \
      __has_builtin(__builtin_amdgcn_s_wait_asynccnt)
#    define USE_ASYNC_LDS 1
#  endif
#endif
#ifndef USE_ASYNC_LDS
#  define USE_ASYNC_LDS 0
#endif

#define AS_GLOBAL_V4(p) ((__attribute__((address_space(1))) v4i*)(p))
#define AS_LOCAL_V4(p)  ((__attribute__((address_space(3))) v4i*)(p))

// ---------------- WMMA fragment helpers (wave32, 16x16x32 f16) ----------------
// A (16x32, MxK): lane L: M = L%16, half = L/16.
//   v16h elements 0..7  = K { 8*half .. 8*half+7 }
//   v16h elements 8..15 = K { 16+8*half .. 16+8*half+7 }
__device__ __forceinline__ v16h frag_a(const _Float16* __restrict__ row, int half) {
  v8h lo = *(const v8h*)(row + 8 * half);
  v8h hi = *(const v8h*)(row + 16 + 8 * half);
  v16h r;
#pragma unroll
  for (int i = 0; i < 8; ++i) { r[i] = lo[i]; r[i + 8] = hi[i]; }
  return r;
}
// B (32x16, KxN): lane L: N = L%16, same K order as frag_a:
//   elements 0..7 = K { 8*half .. +7 }, elements 8..15 = K { 16+8*half .. +7 }.
// Source layout must be [N][K] row-major.
__device__ __forceinline__ v16h frag_b(const _Float16* __restrict__ row, int half) {
  v8h lo = *(const v8h*)(row + 8 * half);
  v8h hi = *(const v8h*)(row + 16 + 8 * half);
  v16h r;
#pragma unroll
  for (int i = 0; i < 8; ++i) { r[i] = lo[i]; r[i + 8] = hi[i]; }
  return r;
}
__device__ __forceinline__ v8f wmma32(v16h a, v16h b, v8f c) {
  return __builtin_amdgcn_wmma_f32_16x16x32_f16(false, a, false, b, (short)0, c, false, false);
}

// ---------------- prep kernels ----------------
__global__ void cvt16(const float* __restrict__ s, _Float16* __restrict__ d, int n) {
  int i = blockIdx.x * 256 + threadIdx.x;
  if (i < n) d[i] = (_Float16)s[i];
}
__global__ void bnconst(const float* __restrict__ g, const float* __restrict__ b,
                        const float* __restrict__ m, const float* __restrict__ v,
                        float* __restrict__ sc, float* __restrict__ bi, int n) {
  int i = blockIdx.x * 256 + threadIdx.x;
  if (i < n) {
    float s = g[i] * rsqrtf(v[i] + 1e-5f);
    sc[i] = s;
    bi[i] = b[i] - m[i] * s;     // also the BN-pad border value
  }
}
__global__ void bnconst2(const float* __restrict__ g2, const float* __restrict__ b2,
                         const float* __restrict__ m2, const float* __restrict__ v2,
                         const float* __restrict__ gq, const float* __restrict__ bq,
                         const float* __restrict__ mq, const float* __restrict__ vq,
                         float* __restrict__ sc, float* __restrict__ bi, int n) {
  int i = blockIdx.x * 256 + threadIdx.x;
  if (i < n) {
    float s2 = g2[i] * rsqrtf(v2[i] + 1e-5f);
    float o2 = b2[i] - m2[i] * s2;
    float sq = gq[i] * rsqrtf(vq[i] + 1e-5f);
    float oq = bq[i] - mq[i] * sq;
    sc[i] = s2 * sq;
    bi[i] = o2 * sq + oq;
  }
}

// ---------------- head LIF (soft reset, tau=2, vth=1) ----------------
// x [T,B,C,196] f32 -> ShT [TB][NP][C] f16 spikes (zero-padded rows n>=196)
__global__ __launch_bounds__(256) void lif_head(const float* __restrict__ x,
                                                _Float16* __restrict__ ShT) {
  __shared__ float sm[16][17];
  int tx = threadIdx.x, ty = threadIdx.y;
  int n_in = blockIdx.x * 16 + tx;
  int c_in = blockIdx.y * 16 + ty;
  int b = blockIdx.z;
  float v = 0.f;
#pragma unroll
  for (int t = 0; t < T; ++t) {
    float xv = (n_in < N) ? x[(((size_t)(t * Bz + b)) * C + c_in) * N + n_in] : 0.f;
    v += (xv - v) * 0.5f;
    float s = (v >= 1.f) ? 1.f : 0.f;
    v -= s;
    sm[ty][tx] = s;
    __syncthreads();
    int n_out = blockIdx.x * 16 + ty;
    int c_out = blockIdx.y * 16 + tx;
    ShT[(((size_t)(t * Bz + b)) * NP + n_out) * C + c_out] = (_Float16)sm[tx][ty];
    __syncthreads();
  }
}

// ---------------- fused 1x1-conv GEMM + BN epilogue ----------------
// D[m][n] = sum_k W[m][k] * Bm[img][n][k]; out = D*sc[m] + bi[m]
// MODE 0: f32 -> Out[img][Cout][NP]
// MODE 1: f16 -> Out[img][Cout][NP]
// MODE 2: f32 -> Out[img][Cout][196] (final output, n guard)
template <int MODE>
__global__ __launch_bounds__(256) void gemm_conv(const _Float16* __restrict__ Wm,
                                                 const _Float16* __restrict__ Bm,
                                                 const float* __restrict__ sc,
                                                 const float* __restrict__ bi,
                                                 void* __restrict__ Out, int Cout) {
  __shared__ __align__(16) _Float16 As[128][40];
  __shared__ __align__(16) _Float16 Bs[64][40];
  const int tid = threadIdx.x;
  const int lane = tid & 31, wave = tid >> 5;
  const int wm = wave >> 1, wn = wave & 1;
  const int half = lane >> 4, l16 = lane & 15;
  const int n0 = blockIdx.x * 64;
  const int m0 = blockIdx.y * 128;
  const int img = blockIdx.z;
  const _Float16* Bimg = Bm + (size_t)img * NP * C;

  v8f acc[2][2];
#pragma unroll
  for (int i = 0; i < 2; ++i)
#pragma unroll
    for (int j = 0; j < 2; ++j) acc[i][j] = (v8f)0.0f;

  for (int kk = 0; kk < C; kk += 32) {
#if USE_ASYNC_LDS
    // CDNA5 async engine: global -> LDS without touching VGPR return path.
#pragma unroll
    for (int i = 0; i < 2; ++i) {
      int id = tid * 2 + i;              // 0..511
      int row = id >> 2, c16 = id & 3;   // 128 rows x 4 chunks of 16B
      __builtin_amdgcn_global_load_async_to_lds_b128(
          AS_GLOBAL_V4(Wm + (size_t)(m0 + row) * C + kk + c16 * 8),
          AS_LOCAL_V4(&As[row][c16 * 8]), 0, 0);
    }
    {
      int row = tid >> 2, c16 = tid & 3; // 64 rows x 4 chunks
      __builtin_amdgcn_global_load_async_to_lds_b128(
          AS_GLOBAL_V4(Bimg + (size_t)(n0 + row) * C + kk + c16 * 8),
          AS_LOCAL_V4(&Bs[row][c16 * 8]), 0, 0);
    }
#else
#pragma unroll
    for (int i = 0; i < 2; ++i) {
      int id = tid * 2 + i;              // 0..511
      int row = id >> 2, c16 = id & 3;   // 128 rows x 4 chunks of 16B
      *(uint4*)&As[row][c16 * 8] =
          *(const uint4*)(Wm + (size_t)(m0 + row) * C + kk + c16 * 8);
    }
    {
      int row = tid >> 2, c16 = tid & 3; // 64 rows x 4 chunks
      *(uint4*)&Bs[row][c16 * 8] =
          *(const uint4*)(Bimg + (size_t)(n0 + row) * C + kk + c16 * 8);
    }
#endif
    if (kk + 32 < C) {                   // prefetch next K tile
      __builtin_prefetch(Wm + (size_t)(m0 + (tid >> 1)) * C + kk + 32, 0, 0);
      __builtin_prefetch(Bimg + (size_t)(n0 + (tid >> 2)) * C + kk + 32, 0, 0);
    }
#if USE_ASYNC_LDS
    __builtin_amdgcn_s_wait_asynccnt(0); // our wave's copies landed in LDS
#endif
    __syncthreads();                     // all waves' copies visible
    v16h a0 = frag_a(&As[wm * 32 + l16][0], half);
    v16h a1 = frag_a(&As[wm * 32 + 16 + l16][0], half);
    v16h b0 = frag_b(&Bs[wn * 32 + l16][0], half);
    v16h b1 = frag_b(&Bs[wn * 32 + 16 + l16][0], half);
    acc[0][0] = wmma32(a0, b0, acc[0][0]);
    acc[0][1] = wmma32(a0, b1, acc[0][1]);
    acc[1][0] = wmma32(a1, b0, acc[1][0]);
    acc[1][1] = wmma32(a1, b1, acc[1][1]);
    __syncthreads();
  }
#pragma unroll
  for (int mi = 0; mi < 2; ++mi)
#pragma unroll
    for (int ni = 0; ni < 2; ++ni)
#pragma unroll
      for (int r = 0; r < 8; ++r) {
        int m = m0 + wm * 32 + mi * 16 + r + 8 * half;
        int n = n0 + wn * 32 + ni * 16 + l16;
        float val = acc[mi][ni][r] * sc[m] + bi[m];
        if (MODE == 0) ((float*)Out)[((size_t)img * Cout + m) * NP + n] = val;
        if (MODE == 1) ((_Float16*)Out)[((size_t)img * Cout + m) * NP + n] = (_Float16)val;
        if (MODE == 2) { if (n < N) ((float*)Out)[((size_t)img * Cout + m) * N + n] = val; }
      }
}

// ---------------- depthwise 3x3 with implicit BN-pad border ----------------
// Y [img][C][NP] f32 (first 196 valid, BN already applied) -> Zt [img][NP][C] f16
__global__ __launch_bounds__(256) void dwconv(const float* __restrict__ Y,
                                              const float* __restrict__ wdw,
                                              const float* __restrict__ pv,
                                              _Float16* __restrict__ Zt) {
  __shared__ float sm[16][17];
  int tx = threadIdx.x, ty = threadIdx.y;
  int img = blockIdx.z;
  int c = blockIdx.y * 16 + ty;
  int n = blockIdx.x * 16 + tx;
  float acc = 0.f;
  if (n < N) {
    int i = n / 14, j = n % 14;
    const float* Yc = Y + ((size_t)img * C + c) * NP;
    float p = pv[c];
#pragma unroll
    for (int ki = 0; ki < 3; ++ki)
#pragma unroll
      for (int kj = 0; kj < 3; ++kj) {
        int ii = i + ki - 1, jj = j + kj - 1;
        float val = (ii >= 0 && ii < 14 && jj >= 0 && jj < 14) ? Yc[ii * 14 + jj] : p;
        acc += wdw[c * 9 + ki * 3 + kj] * val;
      }
  }
  sm[ty][tx] = acc;
  __syncthreads();
  int c_w = blockIdx.y * 16 + tx;
  int n_w = blockIdx.x * 16 + ty;
  Zt[((size_t)img * NP + n_w) * C + c_w] = (_Float16)sm[tx][ty];
}

// ---------------- q/k/v LIF (soft reset) ----------------
// QKV [tb][3C][NP] f16 -> Qs [tb][NP][C], Ks/Vs [tb][C][NP] f16 spikes
__global__ __launch_bounds__(256) void lif_qkv(const _Float16* __restrict__ QKV,
                                               _Float16* __restrict__ Qs,
                                               _Float16* __restrict__ Ks,
                                               _Float16* __restrict__ Vs) {
  int n = threadIdx.x;   // 0..255
  int c = blockIdx.x;    // 0..511
  int b = blockIdx.y;    // 0..15
  float vq = 0.f, vk = 0.f, vv = 0.f;
#pragma unroll
  for (int t = 0; t < T; ++t) {
    size_t tb = (size_t)(t * Bz + b);
    size_t base = tb * C3 * NP;
    float xq = (n < N) ? (float)QKV[base + (size_t)(c) * NP + n] : 0.f;
    float xk = (n < N) ? (float)QKV[base + (size_t)(C + c) * NP + n] : 0.f;
    float xv = (n < N) ? (float)QKV[base + (size_t)(2 * C + c) * NP + n] : 0.f;
    vq += (xq - vq) * 0.5f; float sq = vq >= 1.f ? 1.f : 0.f; vq -= sq;
    vk += (xk - vk) * 0.5f; float sk = vk >= 1.f ? 1.f : 0.f; vk -= sk;
    vv += (xv - vv) * 0.5f; float sv = vv >= 1.f ? 1.f : 0.f; vv -= sv;
    Qs[(tb * NP + n) * C + c] = (_Float16)sq;
    Ks[(tb * C + c) * NP + n] = (_Float16)sk;
    Vs[(tb * C + c) * NP + n] = (_Float16)sv;
  }
}

// ---------------- attention: kv[d][e] = sum_n k[n][d] v[n][e] ----------------
// Ks,Vs [tb][C][NP]; out kvT [tbh][e][d] f16 with 0.125 folded in
__global__ __launch_bounds__(256) void gemm_kv(const _Float16* __restrict__ Ks,
                                               const _Float16* __restrict__ Vs,
                                               _Float16* __restrict__ kvT) {
  int tbh = blockIdx.x;             // 0..511
  int tb = tbh >> 3, h = tbh & 7;
  int lane = threadIdx.x & 31, wave = threadIdx.x >> 5;
  int mtile = wave & 3, npair = wave >> 2;
  int half = lane >> 4, l16 = lane & 15;
  const _Float16* Kb = Ks + ((size_t)tb * C + h * HD) * NP;
  const _Float16* Vb = Vs + ((size_t)tb * C + h * HD) * NP;
  v8f acc0 = (v8f)0.0f, acc1 = (v8f)0.0f;
  for (int kk = 0; kk < NP; kk += 32) {
    v16h a  = frag_a(Kb + (size_t)(mtile * 16 + l16) * NP + kk, half);
    v16h b0 = frag_b(Vb + (size_t)(npair * 32 + l16) * NP + kk, half);
    v16h b1 = frag_b(Vb + (size_t)(npair * 32 + 16 + l16) * NP + kk, half);
    acc0 = wmma32(a, b0, acc0);
    acc1 = wmma32(a, b1, acc1);
  }
#pragma unroll
  for (int t2 = 0; t2 < 2; ++t2) {
    v8f a = (t2 == 0) ? acc0 : acc1;
    int e = (npair * 2 + t2) * 16 + l16;
    v8h out;
#pragma unroll
    for (int r = 0; r < 8; ++r) out[r] = (_Float16)(a[r] * 0.125f);  // SCALE folded
    *(v8h*)&kvT[((size_t)tbh * HD + e) * HD + mtile * 16 + 8 * half] = out;
  }
}

// ---------------- attention: o[n][e] = sum_d q[n][d] kvT[e][d] ----------------
// Qs [tb][NP][C]; O [tb][NP][C] f32
__global__ __launch_bounds__(256) void gemm_q_kv(const _Float16* __restrict__ Qs,
                                                 const _Float16* __restrict__ kvT,
                                                 float* __restrict__ O) {
  int tbh = blockIdx.y;
  int tb = tbh >> 3, h = tbh & 7;
  int wave = threadIdx.x >> 5, lane = threadIdx.x & 31;
  int mtile = blockIdx.x * 8 + wave;  // 0..15
  int half = lane >> 4, l16 = lane & 15;
  const _Float16* Qb = Qs + (size_t)tb * NP * C + h * HD;
  const _Float16* KVb = kvT + (size_t)tbh * HD * HD;
  v8f acc[4];
#pragma unroll
  for (int i = 0; i < 4; ++i) acc[i] = (v8f)0.0f;
#pragma unroll
  for (int kk = 0; kk < HD; kk += 32) {
    v16h a = frag_a(Qb + (size_t)(mtile * 16 + l16) * C + kk, half);
#pragma unroll
    for (int nt = 0; nt < 4; ++nt) {
      v16h b = frag_b(KVb + (size_t)(nt * 16 + l16) * HD + kk, half);
      acc[nt] = wmma32(a, b, acc[nt]);
    }
  }
#pragma unroll
  for (int nt = 0; nt < 4; ++nt)
#pragma unroll
    for (int r = 0; r < 8; ++r) {
      int n = mtile * 16 + r + 8 * half;
      int c = h * HD + nt * 16 + l16;
      O[((size_t)tb * NP + n) * C + c] = acc[nt][r];
    }
}

// ---------------- attn LIF (hard reset, vth=0.5, v_reset=0) ----------------
// O [tb][NP][C] f32 -> Ot [tb][NP][C] f16 spikes
__global__ __launch_bounds__(512) void attn_lif(const float* __restrict__ O,
                                                _Float16* __restrict__ Ot) {
  int c = threadIdx.x;   // 0..511
  int n = blockIdx.x;    // 0..255
  int b = blockIdx.y;
  float v = 0.f;
#pragma unroll
  for (int t = 0; t < T; ++t) {
    size_t idx = (((size_t)(t * Bz + b)) * NP + n) * C + c;
    float xv = (n < N) ? O[idx] : 0.f;
    v += (xv - v) * 0.5f;
    float s = v >= 0.5f ? 1.f : 0.f;
    v = (1.f - s) * v;
    Ot[idx] = (_Float16)s;
  }
}

// ---------------- host launch ----------------
extern "C" void kernel_launch(void* const* d_in, const int* in_sizes, int n_in,
                              void* d_out, int out_size, void* d_ws, size_t ws_size,
                              hipStream_t stream) {
  (void)in_sizes; (void)n_in; (void)out_size; (void)ws_size;
  // inputs (setup_inputs dict order, nested dicts flattened in insertion order)
  const float* x        = (const float*)d_in[0];
  const float* w1       = (const float*)d_in[1];   // [512,512]
  const float* bn1g     = (const float*)d_in[2];
  const float* bn1b     = (const float*)d_in[3];
  const float* bn1m     = (const float*)d_in[4];
  const float* bn1v     = (const float*)d_in[5];
  const float* wdw      = (const float*)d_in[6];   // [512,1,3,3]
  const float* wpw      = (const float*)d_in[7];   // [1536,512]
  const float* bn2g     = (const float*)d_in[8];
  const float* bn2b     = (const float*)d_in[9];
  const float* bn2m     = (const float*)d_in[10];
  const float* bn2v     = (const float*)d_in[11];
  const float* qbng     = (const float*)d_in[12];
  const float* qbnb     = (const float*)d_in[13];
  const float* qbnm     = (const float*)d_in[14];
  const float* qbnv     = (const float*)d_in[15];
  const float* w1p      = (const float*)d_in[16];  // [512,512]
  const float* bn1pg    = (const float*)d_in[17];
  const float* bn1pb    = (const float*)d_in[18];
  const float* bn1pm    = (const float*)d_in[19];
  const float* bn1pv    = (const float*)d_in[20];
  const float* wdwp     = (const float*)d_in[21];
  const float* wpwp     = (const float*)d_in[22];  // [512,512]
  const float* bn2pg    = (const float*)d_in[23];
  const float* bn2pb    = (const float*)d_in[24];
  const float* bn2pm    = (const float*)d_in[25];
  const float* bn2pv    = (const float*)d_in[26];
  const float* pbng     = (const float*)d_in[27];
  const float* pbnb     = (const float*)d_in[28];
  const float* pbnm     = (const float*)d_in[29];
  const float* pbnv     = (const float*)d_in[30];

  // workspace layout
  char* ws = (char*)d_ws;
  size_t off = 0;
  auto take = [&](size_t bytes) { size_t o = off; off = (off + bytes + 255) & ~(size_t)255; return o; };
  _Float16* W1h   = (_Float16*)(ws + take((size_t)C  * C * 2));
  _Float16* Wpwh  = (_Float16*)(ws + take((size_t)C3 * C * 2));
  _Float16* W1ph  = (_Float16*)(ws + take((size_t)C  * C * 2));
  _Float16* Wpwph = (_Float16*)(ws + take((size_t)C  * C * 2));
  float* sc1   = (float*)(ws + take(C  * 4));
  float* bi1   = (float*)(ws + take(C  * 4));
  float* sc2c  = (float*)(ws + take(C3 * 4));
  float* bi2c  = (float*)(ws + take(C3 * 4));
  float* sc1p  = (float*)(ws + take(C  * 4));
  float* bi1p  = (float*)(ws + take(C  * 4));
  float* sc2pc = (float*)(ws + take(C  * 4));
  float* bi2pc = (float*)(ws + take(C  * 4));
  char* regS = ws + take((size_t)TB * NP * C * 2);    // ShT / Zt / Ot (f16)
  char* regA = ws + take((size_t)TB * C * NP * 4);    // Y1 / Qs / Yp
  char* regQ = ws + take((size_t)TB * C3 * NP * 2);   // QKV (f16) / O (f32, fits)
  char* regK = ws + take((size_t)TB * NP * C * 2);    // Ks / Zpt
  char* regV2 = ws + take((size_t)TB * NP * C * 2);   // Vs
  _Float16* kvT = (_Float16*)(ws + take((size_t)TB * HEADS * HD * HD * 2));

  // ---- prep: weight fp16 conversions + BN constant folding ----
  cvt16<<<(C * C + 255) / 256, 256, 0, stream>>>(w1, W1h, C * C);
  cvt16<<<(C3 * C + 255) / 256, 256, 0, stream>>>(wpw, Wpwh, C3 * C);
  cvt16<<<(C * C + 255) / 256, 256, 0, stream>>>(w1p, W1ph, C * C);
  cvt16<<<(C * C + 255) / 256, 256, 0, stream>>>(wpwp, Wpwph, C * C);
  bnconst<<<(C + 255) / 256, 256, 0, stream>>>(bn1g, bn1b, bn1m, bn1v, sc1, bi1, C);
  bnconst2<<<(C3 + 255) / 256, 256, 0, stream>>>(bn2g, bn2b, bn2m, bn2v,
                                                 qbng, qbnb, qbnm, qbnv, sc2c, bi2c, C3);
  bnconst<<<(C + 255) / 256, 256, 0, stream>>>(bn1pg, bn1pb, bn1pm, bn1pv, sc1p, bi1p, C);
  bnconst2<<<(C + 255) / 256, 256, 0, stream>>>(bn2pg, bn2pb, bn2pm, bn2pv,
                                                pbng, pbnb, pbnm, pbnv, sc2pc, bi2pc, C);

  dim3 b16(16, 16);
  // ---- head LIF -> spike f16 [tb][n][c] ----
  lif_head<<<dim3(NP / 16, C / 16, Bz), b16, 0, stream>>>(x, (_Float16*)regS);
  // ---- qkv RepConv: 1x1 + BN (fused) ----
  gemm_conv<0><<<dim3(NP / 64, C / 128, TB), 256, 0, stream>>>(
      W1h, (const _Float16*)regS, sc1, bi1, regA, C);
  // ---- depthwise 3x3 with BN-pad border, transposed f16 out ----
  dwconv<<<dim3(NP / 16, C / 16, TB), b16, 0, stream>>>(
      (const float*)regA, wdw, bi1, (_Float16*)regS);
  // ---- 1x1 C->3C + BN2 + qkv_bn (fused), f16 out ----
  gemm_conv<1><<<dim3(NP / 64, C3 / 128, TB), 256, 0, stream>>>(
      Wpwh, (const _Float16*)regS, sc2c, bi2c, regQ, C3);
  // ---- q/k/v LIF -> spike layouts for attention ----
  lif_qkv<<<dim3(C, Bz), 256, 0, stream>>>((const _Float16*)regQ, (_Float16*)regA,
                                           (_Float16*)regK, (_Float16*)regV2);
  // ---- kv = k^T v (per t,b,h), 0.125 folded, transposed f16 ----
  gemm_kv<<<TB * HEADS, 256, 0, stream>>>((const _Float16*)regK, (const _Float16*)regV2, kvT);
  // ---- o = q * kv ----
  gemm_q_kv<<<dim3(2, TB * HEADS), 256, 0, stream>>>((const _Float16*)regA, kvT, (float*)regQ);
  // ---- attn LIF (hard reset) ----
  attn_lif<<<dim3(NP, Bz), 512, 0, stream>>>((const float*)regQ, (_Float16*)regS);
  // ---- proj RepConv: 1x1 + BN (fused) ----
  gemm_conv<0><<<dim3(NP / 64, C / 128, TB), 256, 0, stream>>>(
      W1ph, (const _Float16*)regS, sc1p, bi1p, regA, C);
  dwconv<<<dim3(NP / 16, C / 16, TB), b16, 0, stream>>>(
      (const float*)regA, wdwp, bi1p, (_Float16*)regK);
  // ---- 1x1 C->C + BN2 + proj_bn (fused), f32 out to d_out [T,B,C,14,14] ----
  gemm_conv<2><<<dim3(NP / 64, C / 128, TB), 256, 0, stream>>>(
      Wpwph, (const _Float16*)regK, sc2pc, bi2pc, d_out, C);
}
